// SchNetInteraction_31310311587918
// MI455X (gfx1250) — matrix operand order repaired and checked by reading, not compiled, not run on hardware
//
#include <hip/hip_runtime.h>
#include <hip/hip_bf16.h>
#include <math.h>

#define C 128           // channels
#define F 50            // gaussian expansion
#define KF 64           // F padded to WMMA K multiple (32)
#define CUTOFF 5.0f
#define LOG2F_ 0.6931471805599453f

typedef __bf16 bf16_t;
typedef __attribute__((ext_vector_type(16))) __bf16 v16bf;
typedef __attribute__((ext_vector_type(8)))  float  v8f;

// ---------------------------------------------------------------------------
// helpers
// ---------------------------------------------------------------------------
__device__ __forceinline__ bf16_t f2bf(float x) { return (bf16_t)x; }

__device__ __forceinline__ void store_bf16x4(bf16_t* dst, float4 v) {
    union { bf16_t b[4]; uint2 u; } t;
    t.b[0] = f2bf(v.x); t.b[1] = f2bf(v.y); t.b[2] = f2bf(v.z); t.b[3] = f2bf(v.w);
    *(uint2*)dst = t.u;
}

__device__ __forceinline__ float ssp(float x) {
    // shifted softplus: log(1+e^x) - log(2), numerically stable
    return fmaxf(x, 0.f) + __logf(1.f + __expf(-fabsf(x))) - LOG2F_;
}

// A-matrix fragment, 16x32 bf16 (ISA 7.12.2):
//   lanes 0-15 : row M=lane,    VGPR0-3 = K[koff+0..7],  VGPR4-7 = K[koff+16..23]
//   lanes 16-31: row M=lane-16, VGPR0-3 = K[koff+8..15], VGPR4-7 = K[koff+24..31]
__device__ __forceinline__ v16bf load_frag_a(const bf16_t* lds, int stride,
                                             int koff, int lane) {
    int m  = lane & 15;
    int kb = (lane & 16) ? 8 : 0;
    const char* base = (const char*)(lds + m * stride + koff);
    union { v16bf v; uint4 u[2]; } r;
    r.u[0] = *(const uint4*)(base + (size_t)(kb) * 2);
    r.u[1] = *(const uint4*)(base + (size_t)(16 + kb) * 2);
    return r.v;
}

// B-matrix fragment, 32x16 bf16: lane L holds column N=nt*16+(L&15); elements
// 0-7 = K[koff+kb+0..7], 8-15 = K[koff+16+kb+0..7], kb = (L&16) ? 8 : 0.
__device__ __forceinline__ v16bf load_frag_b(const bf16_t* lds, int stride,
                                             int koff, int nt, int lane) {
    int kb = (lane & 16) ? 8 : 0;
    const bf16_t* base = lds + (size_t)(koff + kb) * stride + nt * 16 + (lane & 15);
    v16bf r;
#pragma unroll
    for (int i = 0; i < 8; ++i) {
        r[i]     = base[(size_t)i * stride];
        r[8 + i] = base[(size_t)(16 + i) * stride];
    }
    return r;
}

__device__ __forceinline__ v8f wmma_bf16(v16bf a, v16bf b, v8f c) {
    return __builtin_amdgcn_wmma_f32_16x16x32_bf16(
        false, a, false, b, (short)0, c, false, false);
}

// ---------------------------------------------------------------------------
// 0) zero the scatter accumulator
// ---------------------------------------------------------------------------
__global__ void zero_agg(float4* __restrict__ p, int n4) {
    int i = blockIdx.x * blockDim.x + threadIdx.x;
    if (i < n4) p[i] = make_float4(0.f, 0.f, 0.f, 0.f);
}

// ---------------------------------------------------------------------------
// 1) h = features @ W_lin^T   [N,C] = [N,C] x [C,C]^T
// ---------------------------------------------------------------------------
__global__ void node_linear(const float* __restrict__ feat,
                            const float* __restrict__ W,
                            float* __restrict__ h, int nNodes) {
    __shared__ __align__(16) bf16_t sW[C * C];      // transposed: sW[k*C+c] = W[c*C+k]
    __shared__ __align__(16) bf16_t sX[8][16 * C];

    int tid = threadIdx.x;
    for (int i = tid; i < C * C; i += 256) {        // transpose: scalar, once per block
        int k = i >> 7, c = i & 127;
        sW[i] = f2bf(W[c * C + k]);
    }
    __syncthreads();

    int wave = tid >> 5, lane = tid & 31;
    int nTiles = (nNodes + 15) >> 4;
    for (int tile = blockIdx.x * 8 + wave; tile < nTiles; tile += gridDim.x * 8) {
        int n0 = tile << 4;
        bool full = (n0 + 16) <= nNodes;
        if (full) {
            const float4* src = (const float4*)(feat + (size_t)n0 * C);
            for (int i = lane; i < 16 * C / 4; i += 32)
                store_bf16x4(&sX[wave][i * 4], src[i]);
        } else {
            for (int i = lane; i < 16 * C; i += 32) {
                int r = i >> 7, c = i & 127;
                int node = n0 + r;
                sX[wave][i] = f2bf(node < nNodes ? feat[(size_t)node * C + c] : 0.f);
            }
        }

        v8f acc[8] = {};
#pragma unroll
        for (int kt = 0; kt < 4; ++kt) {
            v16bf a = load_frag_a(sX[wave], C, kt * 32, lane);
#pragma unroll
            for (int nt = 0; nt < 8; ++nt) {
                v16bf b = load_frag_b(sW, C, kt * 32, nt, lane);
                acc[nt] = wmma_bf16(a, b, acc[nt]);
            }
        }

        int col0 = lane & 15;
        int mbase = (lane >> 4) << 3;
        if (full) {
#pragma unroll
            for (int nt = 0; nt < 8; ++nt) {
                float* p = h + (size_t)(n0 + mbase) * C + nt * 16 + col0;
#pragma unroll
                for (int i = 0; i < 8; ++i) p[i * C] = acc[nt][i];
            }
        } else {
#pragma unroll
            for (int nt = 0; nt < 8; ++nt) {
#pragma unroll
                for (int i = 0; i < 8; ++i) {
                    int node = n0 + mbase + i;
                    if (node < nNodes) h[(size_t)node * C + nt * 16 + col0] = acc[nt][i];
                }
            }
        }
    }
}

// ---------------------------------------------------------------------------
// 2) fused edge kernel: basis -> MLP (2 GEMMs) -> gather*mul -> atomic scatter
// ---------------------------------------------------------------------------
__global__ void edge_fused(const float* __restrict__ dist,
                           const float* __restrict__ Wf1, const float* __restrict__ bf1,
                           const float* __restrict__ Wf2, const float* __restrict__ bf2,
                           const int* __restrict__ senders,
                           const int* __restrict__ receivers,
                           const float* __restrict__ h,
                           float* __restrict__ agg, int nEdges) {
    __shared__ __align__(16) bf16_t sWf1[KF * C];     // K padded 50->64 with zeros
    __shared__ __align__(16) bf16_t sWf2[C * C];
    __shared__ float sb1[C], sb2[C];
    __shared__ __align__(16) bf16_t sBasis[8][16 * KF];
    __shared__ __align__(16) bf16_t sAct[8][16 * C];  // act tile, later reused for ef tile
    __shared__ float sDist[8][16];
    __shared__ int   sSend[8][16];
    __shared__ int   sRecv[8][16];

    int tid = threadIdx.x;
    {   // Wf1 [F,C] -> bf16, pad rows F..KF-1 with zeros (vectorized)
        const float4* w1 = (const float4*)Wf1;
        for (int i = tid; i < KF * C / 4; i += 256) {
            int k = (i * 4) >> 7;
            if (k < F) store_bf16x4(&sWf1[i * 4], w1[i]);
            else       *(uint2*)&sWf1[i * 4] = make_uint2(0u, 0u);
        }
        const float4* w2 = (const float4*)Wf2;
        for (int i = tid; i < C * C / 4; i += 256) store_bf16x4(&sWf2[i * 4], w2[i]);
        if (tid < C) { sb1[tid] = bf1[tid]; sb2[tid] = bf2[tid]; }
    }
    __syncthreads();

    const float delta = CUTOFF / (float)(F - 1);
    const float coef  = 1.f / (2.f * delta * delta);

    int wave = tid >> 5, lane = tid & 31;
    int col0 = lane & 15;
    int mbase = (lane >> 4) << 3;
    int nTiles = (nEdges + 15) >> 4;
    for (int tile = blockIdx.x * 8 + wave; tile < nTiles; tile += gridDim.x * 8) {
        int e0 = tile << 4;
        bool full = (e0 + 16) <= nEdges;
        if (lane < 16) {
            int e = e0 + lane;
            bool v = full || (e < nEdges);
            sDist[wave][lane] = v ? dist[e] : 0.f;
            sSend[wave][lane] = v ? senders[e] : 0;
            sRecv[wave][lane] = v ? receivers[e] : -1;
        }
        // Gaussian basis tile [16 x 64] (cols >= 50 zero-padded); same-wave LDS
        // ops are in order, no barrier needed.
        for (int i = lane; i < 16 * KF; i += 32) {
            int el = i >> 6, j = i & 63;
            float val = 0.f;
            if (j < F) {
                float d = sDist[wave][el] - (float)j * delta;
                val = __expf(-coef * d * d);
            }
            sBasis[wave][i] = f2bf(val);
        }

        // GEMM1: [16,64] x [64,128] -> bias -> shifted softplus -> bf16 act
        {
            v8f acc[8] = {};
#pragma unroll
            for (int kt = 0; kt < 2; ++kt) {
                v16bf a = load_frag_a(sBasis[wave], KF, kt * 32, lane);
#pragma unroll
                for (int nt = 0; nt < 8; ++nt) {
                    v16bf b = load_frag_b(sWf1, C, kt * 32, nt, lane);
                    acc[nt] = wmma_bf16(a, b, acc[nt]);
                }
            }
#pragma unroll
            for (int nt = 0; nt < 8; ++nt) {
                int n = nt * 16 + col0;
                float bias = sb1[n];
                bf16_t* p = &sAct[wave][mbase * C + n];
#pragma unroll
                for (int i = 0; i < 8; ++i) p[i * C] = f2bf(ssp(acc[nt][i] + bias));
            }
        }

        // GEMM2: [16,128] x [128,128] -> bias -> ef tile back to LDS (row-major
        // consume) -> coalesced gather of h[sender] -> atomic scatter to agg.
        {
            v8f acc[8] = {};
#pragma unroll
            for (int kt = 0; kt < 4; ++kt) {
                v16bf a = load_frag_a(sAct[wave], C, kt * 32, lane);
#pragma unroll
                for (int nt = 0; nt < 8; ++nt) {
                    v16bf b = load_frag_b(sWf2, C, kt * 32, nt, lane);
                    acc[nt] = wmma_bf16(a, b, acc[nt]);
                }
            }
            // sAct A-fragment reads are done; reuse the buffer for the ef tile.
#pragma unroll
            for (int nt = 0; nt < 8; ++nt) {
                int n = nt * 16 + col0;
                float bias = sb2[n];
                bf16_t* p = &sAct[wave][mbase * C + n];
#pragma unroll
                for (int i = 0; i < 8; ++i) p[i * C] = f2bf(acc[nt][i] + bias);
            }
            // Row-major consume: lane owns edge-row (lane>>1), 64 contiguous
            // channels. 16x global_load_b128 gather, immediate-offset atomics.
            int row = lane >> 1;
            int cb  = (lane & 1) << 6;
            int rcv1 = sRecv[wave][row];
            if (rcv1 >= 0) {
                int snd1 = sSend[wave][row];
                const float4*  hp = (const float4*)(h + (size_t)snd1 * C + cb);
                float*         ap = agg + (size_t)rcv1 * C + cb;
                const bf16_t*  ep = &sAct[wave][row * C + cb];
#pragma unroll
                for (int q = 0; q < 16; ++q) {
                    float4 hv = hp[q];
                    union { uint2 u; bf16_t b[4]; } t;
                    t.u = *(const uint2*)(ep + q * 4);
                    atomicAdd(&ap[q * 4 + 0], (float)t.b[0] * hv.x);
                    atomicAdd(&ap[q * 4 + 1], (float)t.b[1] * hv.y);
                    atomicAdd(&ap[q * 4 + 2], (float)t.b[2] * hv.z);
                    atomicAdd(&ap[q * 4 + 3], (float)t.b[3] * hv.w);
                }
            }
        }
    }
}

// ---------------------------------------------------------------------------
// 3) out = ssp(agg@Wm1+bm1) @ Wm2 + bm2
// ---------------------------------------------------------------------------
__global__ void out_mlp(const float* __restrict__ agg,
                        const float* __restrict__ Wm1, const float* __restrict__ bm1,
                        const float* __restrict__ Wm2, const float* __restrict__ bm2,
                        float* __restrict__ out, int nNodes) {
    __shared__ __align__(16) bf16_t sW1[C * C];
    __shared__ __align__(16) bf16_t sW2[C * C];
    __shared__ float sb1[C], sb2[C];
    __shared__ __align__(16) bf16_t sX[8][16 * C];
    __shared__ __align__(16) bf16_t sT[8][16 * C];

    int tid = threadIdx.x;
    {
        const float4* w1 = (const float4*)Wm1;
        const float4* w2 = (const float4*)Wm2;
        for (int i = tid; i < C * C / 4; i += 256) {
            store_bf16x4(&sW1[i * 4], w1[i]);
            store_bf16x4(&sW2[i * 4], w2[i]);
        }
        if (tid < C) { sb1[tid] = bm1[tid]; sb2[tid] = bm2[tid]; }
    }
    __syncthreads();

    int wave = tid >> 5, lane = tid & 31;
    int col0 = lane & 15;
    int mbase = (lane >> 4) << 3;
    int nTiles = (nNodes + 15) >> 4;
    for (int tile = blockIdx.x * 8 + wave; tile < nTiles; tile += gridDim.x * 8) {
        int n0 = tile << 4;
        bool full = (n0 + 16) <= nNodes;
        if (full) {
            const float4* src = (const float4*)(agg + (size_t)n0 * C);
            for (int i = lane; i < 16 * C / 4; i += 32)
                store_bf16x4(&sX[wave][i * 4], src[i]);
        } else {
            for (int i = lane; i < 16 * C; i += 32) {
                int r = i >> 7, c = i & 127;
                int node = n0 + r;
                sX[wave][i] = f2bf(node < nNodes ? agg[(size_t)node * C + c] : 0.f);
            }
        }
        // layer 1
        {
            v8f acc[8] = {};
#pragma unroll
            for (int kt = 0; kt < 4; ++kt) {
                v16bf a = load_frag_a(sX[wave], C, kt * 32, lane);
#pragma unroll
                for (int nt = 0; nt < 8; ++nt) {
                    v16bf b = load_frag_b(sW1, C, kt * 32, nt, lane);
                    acc[nt] = wmma_bf16(a, b, acc[nt]);
                }
            }
#pragma unroll
            for (int nt = 0; nt < 8; ++nt) {
                int n = nt * 16 + col0;
                float bias = sb1[n];
                bf16_t* p = &sT[wave][mbase * C + n];
#pragma unroll
                for (int i = 0; i < 8; ++i) p[i * C] = f2bf(ssp(acc[nt][i] + bias));
            }
        }
        // layer 2
        {
            v8f acc[8] = {};
#pragma unroll
            for (int kt = 0; kt < 4; ++kt) {
                v16bf a = load_frag_a(sT[wave], C, kt * 32, lane);
#pragma unroll
                for (int nt = 0; nt < 8; ++nt) {
                    v16bf b = load_frag_b(sW2, C, kt * 32, nt, lane);
                    acc[nt] = wmma_bf16(a, b, acc[nt]);
                }
            }
            if (full) {
#pragma unroll
                for (int nt = 0; nt < 8; ++nt) {
                    int n = nt * 16 + col0;
                    float bias = sb2[n];
                    float* p = out + (size_t)(n0 + mbase) * C + n;
#pragma unroll
                    for (int i = 0; i < 8; ++i) p[i * C] = acc[nt][i] + bias;
                }
            } else {
#pragma unroll
                for (int nt = 0; nt < 8; ++nt) {
                    int n = nt * 16 + col0;
                    float bias = sb2[n];
#pragma unroll
                    for (int i = 0; i < 8; ++i) {
                        int node = n0 + mbase + i;
                        if (node < nNodes) out[(size_t)node * C + n] = acc[nt][i] + bias;
                    }
                }
            }
        }
    }
}

// ---------------------------------------------------------------------------
extern "C" void kernel_launch(void* const* d_in, const int* in_sizes, int n_in,
                              void* d_out, int out_size, void* d_ws, size_t ws_size,
                              hipStream_t stream) {
    const float* features = (const float*)d_in[0];
    const float* dist     = (const float*)d_in[1];
    const float* W_lin    = (const float*)d_in[2];
    const float* Wf1      = (const float*)d_in[3];
    const float* bf1      = (const float*)d_in[4];
    const float* Wf2      = (const float*)d_in[5];
    const float* bf2      = (const float*)d_in[6];
    const float* Wm1      = (const float*)d_in[7];
    const float* bm1      = (const float*)d_in[8];
    const float* Wm2      = (const float*)d_in[9];
    const float* bm2      = (const float*)d_in[10];
    const int*   senders  = (const int*)d_in[11];
    const int*   receivers= (const int*)d_in[12];

    int nNodes = in_sizes[0] / C;
    int nEdges = in_sizes[1];

    float* h   = (float*)d_ws;                       // [N, C]
    float* agg = h + (size_t)nNodes * C;             // [N, C]
    float* out = (float*)d_out;

    {   // 0) zero agg
        int n4 = (nNodes * C) / 4;
        zero_agg<<<(n4 + 255) / 256, 256, 0, stream>>>((float4*)agg, n4);
    }
    {   // 1) h = X @ W_lin^T
        int nTiles = (nNodes + 15) / 16;
        int blocks = (nTiles + 7) / 8;
        node_linear<<<blocks, 256, 0, stream>>>(features, W_lin, h, nNodes);
    }
    {   // 2) fused edge pass (compute-dominant)
        int nTiles = (nEdges + 15) / 16;
        int blocks = (nTiles + 7) / 8;
        if (blocks > 1024) blocks = 1024;            // grid-stride, amortize weight staging
        edge_fused<<<blocks, 256, 0, stream>>>(dist, Wf1, bf1, Wf2, bf2,
                                               senders, receivers, h, agg, nEdges);
    }
    {   // 3) output MLP
        int nTiles = (nNodes + 15) / 16;
        int blocks = (nTiles + 7) / 8;
        out_mlp<<<blocks, 256, 0, stream>>>(agg, Wm1, bm1, Wm2, bm2, out, nNodes);
    }
}
